// TransformerAttentionLayer_61598420959309
// MI455X (gfx1250) — matrix operand order repaired and checked by reading, not compiled, not run on hardware
//
#include <hip/hip_runtime.h>

#define N_DST  32768
#define N_EDGE 524288
#define D_NODE 128
#define D_EDGE 128
#define D_TIME 100
#define D_OUT  128

// GEMM K dims
#define KV_K   384   // 128 + 128 + 100 -> pad to 384 (12 chunks of 32)
#define Q_K    128   // time-encoding part folded into constant bias
#define OUT_K  256   // 128 + 128 (8 chunks of 32)
#define KV_N   256   // K outputs (128) + V outputs (128)

typedef unsigned short u16;
typedef unsigned int   u32;

typedef __attribute__((ext_vector_type(16))) __bf16 v16bf;
typedef __attribute__((ext_vector_type(2)))  __bf16 v2bf;
typedef __attribute__((ext_vector_type(2)))  float  v2f;
typedef __attribute__((ext_vector_type(8)))  float  v8f;

struct __align__(16) B128 { u32 u[4]; };
union Frag16 { v16bf v; B128 q[2]; };

// ---------- helpers (native bf16 converts) ----------
__device__ __forceinline__ u16 bfbits(float f) {
    union { __bf16 h; u16 u; } x; x.h = (__bf16)f; return x.u;
}
// single v_cvt_pk_bf16_f32: two floats -> packed bf16 pair
__device__ __forceinline__ u32 pack2bf(float a, float b) {
    v2f f; f.x = a; f.y = b;
    v2bf h = __builtin_convertvector(f, v2bf);
    union { v2bf v; u32 u; } r; r.v = h; return r.u;
}
__device__ __forceinline__ float bf2f(u16 h) {
    union { u32 u; float f; } x; x.u = ((u32)h) << 16; return x.f;
}
__device__ __forceinline__ void load16(const float* __restrict__ p, float* vals) {
    const float4* s = (const float4*)p;
    float4 a = s[0], b = s[1], c = s[2], d = s[3];
    vals[0]  = a.x; vals[1]  = a.y; vals[2]  = a.z; vals[3]  = a.w;
    vals[4]  = b.x; vals[5]  = b.y; vals[6]  = b.z; vals[7]  = b.w;
    vals[8]  = c.x; vals[9]  = c.y; vals[10] = c.z; vals[11] = c.w;
    vals[12] = d.x; vals[13] = d.y; vals[14] = d.z; vals[15] = d.w;
}
__device__ __forceinline__ void packStore(u16* dst, const float* vals) {
    u32* d = (u32*)dst;
#pragma unroll
    for (int c = 0; c < 8; ++c) d[c] = pack2bf(vals[2 * c], vals[2 * c + 1]);
}

// Workspace layout (bytes)
#define OFF_WKV    0ull           // 256*384*2      = 196608
#define OFF_WQ     196608ull      // 128*128*2      = 32768
#define OFF_WOUT   229376ull      // 128*256*2      = 65536
#define OFF_QCONST 294912ull      // 128*4          = 512
#define OFF_ROWP   295424ull      // (N_DST+1)*4    = 131076
#define OFF_Q      524288ull      // N_DST*128*4    = 16777216
#define OFF_AGG    17301504ull    // N_DST*128*4    = 16777216
#define OFF_KV     34078720ull    // N_EDGE*256*2   = 268435456  (total ~302.5 MB)

// ---------- kernel 0: pack weights to bf16 ----------
__global__ __launch_bounds__(256) void prep_weights(
    const float* __restrict__ w_q_w, const float* __restrict__ w_k_w,
    const float* __restrict__ w_v_w, const float* __restrict__ w_out_w,
    u16* __restrict__ wQ, u16* __restrict__ wKV, u16* __restrict__ wOut)
{
    int idx = blockIdx.x * 256 + threadIdx.x;          // 576 blocks -> 147456 threads
    if (idx < 256 * KV_K) {                            // wKV [256][384]
        int n = idx / KV_K, k = idx % KV_K;
        float v = 0.f;
        if (k < 356) v = (n < 128) ? w_k_w[n * 356 + k] : w_v_w[(n - 128) * 356 + k];
        wKV[idx] = bfbits(v);
    } else if (idx < 256 * KV_K + 128 * Q_K) {         // wQ [128][128] (node part only)
        int i = idx - 256 * KV_K;
        int n = i / Q_K, k = i % Q_K;
        wQ[i] = bfbits(w_q_w[n * 228 + k]);
    } else {                                           // wOut [128][256]
        int i = idx - (256 * KV_K + 128 * Q_K);
        wOut[i] = bfbits(w_out_w[i]);
    }
}

// qconst[n] = sum_j cos(time_b[j]) * w_q_w[n][128+j]  (row-independent Q contribution)
__global__ void prep_qconst(const float* __restrict__ w_q_w,
                            const float* __restrict__ time_b, float* __restrict__ qc)
{
    int n = threadIdx.x;            // 128 threads
    float s = 0.f;
    for (int j = 0; j < D_TIME; ++j)
        s += __cosf(time_b[j]) * w_q_w[n * 228 + 128 + j];
    qc[n] = s;
}

// ---------- kernel 1: CSR row pointers from sorted edge_dst ----------
__global__ __launch_bounds__(256) void build_rowptr(
    const int* __restrict__ edge_dst, int* __restrict__ rowp)
{
    int e = blockIdx.x * 256 + threadIdx.x;
    if (e >= N_EDGE) return;
    int d = edge_dst[e];
    int prev = (e == 0) ? -1 : edge_dst[e - 1];
    for (int x = prev + 1; x <= d; ++x) rowp[x] = e;
    if (e == N_EDGE - 1)
        for (int x = d + 1; x <= N_DST; ++x) rowp[x] = N_EDGE;
}

// ---------- fragment loaders ----------
__device__ __forceinline__ Frag16 loadA(const u16* As /*[rows][40]*/, int rowBase, int lane) {
    Frag16 a;
    int rr = rowBase + (lane & 15);
    int hh = (lane >> 4) * 8;
    const u16* p = As + rr * 40;
    a.q[0] = *(const B128*)(p + hh);
    a.q[1] = *(const B128*)(p + 16 + hh);
    return a;
}
__device__ __forceinline__ Frag16 loadB(const u16* Bs, int n0, int lane) {
    Frag16 b;
    const u16* p = Bs + (n0 + (lane & 15)) * 40 + (lane >> 4) * 8;
    b.q[0] = *(const B128*)p;
    b.q[1] = *(const B128*)(p + 16);
    return b;
}
__device__ __forceinline__ void stageB(const u16* __restrict__ W, int pitch, int rows,
                                       int k0, u16* Bs, int tid) {
    int part = tid & 3;                      // 16B each, 4 threads/row
#pragma unroll
    for (int r = tid >> 2; r < rows; r += 64) {
        const B128* src = (const B128*)(W + (size_t)r * pitch + k0 + part * 8);
        *(B128*)(Bs + r * 40 + part * 8) = *src;
    }
}

// ---------- kernel 2: Q = h_src[:N] @ wQ^T + (qb + qconst) -> fp32 [N_DST][128] ----------
__global__ __launch_bounds__(256) void gemm_q(
    const float* __restrict__ h_src, const u16* __restrict__ wQ,
    const float* __restrict__ qb, const float* __restrict__ qc, float* __restrict__ Qout)
{
    __shared__ u16 As[128][40];
    __shared__ u16 Bs[128][40];
    int mBase = blockIdx.x * 128;
    int tid = threadIdx.x, wv = tid >> 5, lane = tid & 31;
    int r = tid >> 1, cb = (tid & 1) * 16;
    const float* pA = h_src + (size_t)(mBase + r) * 128 + cb;
    v8f acc[8];
    v8f z = {};
#pragma unroll
    for (int t = 0; t < 8; ++t) acc[t] = z;

    for (int kc = 0; kc < Q_K / 32; ++kc) {
        int k0 = kc * 32;
        float vals[16];
        load16(pA + k0, vals);
        packStore(&As[r][cb], vals);
        stageB(wQ, Q_K, 128, k0, &Bs[0][0], tid);
        __syncthreads();
        Frag16 a = loadA(&As[0][0], wv * 16, lane);
        Frag16 bf = loadB(&Bs[0][0], 0, lane);
#pragma unroll
        for (int t = 0; t < 8; ++t) {
            Frag16 bn;
            if (t + 1 < 8) bn = loadB(&Bs[0][0], (t + 1) * 16, lane);
            acc[t] = __builtin_amdgcn_wmma_f32_16x16x32_bf16(
                false, a.v, false, bf.v, (short)0, acc[t], false, false);
            if (t + 1 < 8) bf = bn;
        }
        __syncthreads();
    }
#pragma unroll
    for (int t = 0; t < 8; ++t) {
        int col = t * 16 + (lane & 15);
        float bias = qb[col] + qc[col];
#pragma unroll
        for (int r8 = 0; r8 < 8; ++r8) {
            int row = mBase + wv * 16 + r8 + 8 * (lane >> 4);
            Qout[(size_t)row * 128 + col] = acc[t][r8] + bias;
        }
    }
}

// ---------- kernel 3: KV = [h_src[N:] | edge_feat | cos(dt*w+b)] @ wKV^T + bias -> bf16 [E][256] ----------
__global__ __launch_bounds__(256) void gemm_kv(
    const float* __restrict__ h_src, const float* __restrict__ edge_feat,
    const float* __restrict__ dt, const float* __restrict__ time_w,
    const float* __restrict__ time_b, const u16* __restrict__ wKV,
    const float* __restrict__ kb, const float* __restrict__ vb, u16* __restrict__ KV)
{
    __shared__ u16 As[2][128][40];    // 20 KB
    __shared__ u16 Bs[2][256][40];    // 40 KB
    int mBase = blockIdx.x * 128;
    int tid = threadIdx.x, wv = tid >> 5, lane = tid & 31;
    int rA = tid >> 1, cbA = (tid & 1) * 16, e = mBase + rA;
    const float* pH = h_src + (size_t)(N_DST + e) * 128 + cbA;
    const float* pE = edge_feat + (size_t)e * 128 + cbA;
    float tval = dt[e];

    v8f acc[16];
    v8f z = {};
#pragma unroll
    for (int t = 0; t < 16; ++t) acc[t] = z;

    auto stage = [&](int kc, int buf) {
        int kbv = kc * 32 + cbA;     // multiple of 16; regions 0/128/256 are 16-aligned
        float vals[16];
        if (kbv < 128) {
            load16(pH + kc * 32, vals);
        } else if (kbv < 256) {
            load16(pE + kc * 32 - 128, vals);
        } else {
#pragma unroll
            for (int c = 0; c < 16; ++c) {
                int k = kbv + c;
                vals[c] = (k < 356)
                    ? __cosf(tval * time_w[k - 256] + time_b[k - 256]) : 0.f;
            }
        }
        packStore(&As[buf][rA][cbA], vals);
        stageB(wKV, KV_K, 256, kc * 32, &Bs[buf][0][0], tid);
    };

    stage(0, 0);
    __syncthreads();
    int buf = 0;
    for (int kc = 0; kc < KV_K / 32; ++kc) {
        if (kc + 1 < KV_K / 32) stage(kc + 1, buf ^ 1);
        Frag16 a = loadA(&As[buf][0][0], wv * 16, lane);
        Frag16 bf = loadB(&Bs[buf][0][0], 0, lane);
#pragma unroll
        for (int t = 0; t < 16; ++t) {
            Frag16 bn;
            if (t + 1 < 16) bn = loadB(&Bs[buf][0][0], (t + 1) * 16, lane);
            acc[t] = __builtin_amdgcn_wmma_f32_16x16x32_bf16(
                false, a.v, false, bf.v, (short)0, acc[t], false, false);
            if (t + 1 < 16) bf = bn;
        }
        __syncthreads();
        buf ^= 1;
    }
#pragma unroll
    for (int t = 0; t < 16; ++t) {
        int col = t * 16 + (lane & 15);
        float bias = (col < 128) ? kb[col] : vb[col - 128];
#pragma unroll
        for (int r8 = 0; r8 < 8; ++r8) {
            int row = mBase + wv * 16 + r8 + 8 * (lane >> 4);
            KV[(size_t)row * KV_N + col] = bfbits(acc[t][r8] + bias);
        }
    }
}

// ---------- kernel 4: per-destination online-softmax attention (one wave per dest) ----------
__global__ __launch_bounds__(256) void attn(
    const float* __restrict__ Q, const u16* __restrict__ KV,
    const int* __restrict__ rowp, float* __restrict__ agg)
{
    int d = blockIdx.x * 8 + (threadIdx.x >> 5);    // 4096 blocks * 8 waves = 32768
    int lane = threadIdx.x & 31;
    // lane holds dims [lane*4, lane*4+4): lanes 0-15 = head0, 16-31 = head1
    float q0, q1, q2, q3;
    {
        const float4 qv = *(const float4*)(Q + (size_t)d * 128 + lane * 4);
        q0 = qv.x; q1 = qv.y; q2 = qv.z; q3 = qv.w;
    }
    int beg = rowp[d], end = rowp[d + 1];
    float m = -3.0e38f, sum = 0.f;
    float a0 = 0.f, a1 = 0.f, a2 = 0.f, a3 = 0.f;
    for (int e = beg; e < end; ++e) {
        const u16* row = KV + (size_t)e * KV_N;
        if (e + 1 < N_EDGE)
            __builtin_prefetch(row + KV_N + lane * 8, 0, 3);   // next edge's K/V row
        unsigned long long kk = *(const unsigned long long*)(row + lane * 4);
        float p = q0 * bf2f((u16)(kk       )) + q1 * bf2f((u16)(kk >> 16))
                + q2 * bf2f((u16)(kk >> 32)) + q3 * bf2f((u16)(kk >> 48));
        // reduce over the 16 lanes of this head (wave32 halves stay disjoint)
        p += __shfl_xor(p, 1, 32);
        p += __shfl_xor(p, 2, 32);
        p += __shfl_xor(p, 4, 32);
        p += __shfl_xor(p, 8, 32);
        float s = (p > 0.f) ? p : 0.2f * p;           // leaky_relu(·, 0.2)
        float mn = fmaxf(m, s);
        float corr = __expf(m - mn);
        float w = __expf(s - mn);
        sum = sum * corr + w;
        unsigned long long vv = *(const unsigned long long*)(row + 128 + lane * 4);
        a0 = a0 * corr + w * bf2f((u16)(vv       ));
        a1 = a1 * corr + w * bf2f((u16)(vv >> 16));
        a2 = a2 * corr + w * bf2f((u16)(vv >> 32));
        a3 = a3 * corr + w * bf2f((u16)(vv >> 48));
        m = mn;
    }
    float inv = (sum > 0.f) ? (1.f / sum) : 0.f;      // empty segment -> zeros
    float* op = agg + (size_t)d * 128 + lane * 4;
    op[0] = a0 * inv; op[1] = a1 * inv; op[2] = a2 * inv; op[3] = a3 * inv;
}

// ---------- kernel 5: out = LN(relu([agg | h_src[:N]] @ wOut^T + b)) ----------
__global__ __launch_bounds__(256) void gemm_out(
    const float* __restrict__ agg, const float* __restrict__ h_src,
    const u16* __restrict__ wOut, const float* __restrict__ ob,
    const float* __restrict__ g, const float* __restrict__ bb, float* __restrict__ out)
{
    __shared__ u16 As[128][40];
    __shared__ u16 Bs[128][40];
    int mBase = blockIdx.x * 128;
    int tid = threadIdx.x, wv = tid >> 5, lane = tid & 31;
    int r = tid >> 1, cb = (tid & 1) * 16;
    const float* pAgg = agg + (size_t)(mBase + r) * 128 + cb;
    const float* pH   = h_src + (size_t)(mBase + r) * 128 + cb;
    v8f acc[8];
    v8f z = {};
#pragma unroll
    for (int t = 0; t < 8; ++t) acc[t] = z;

    for (int kc = 0; kc < OUT_K / 32; ++kc) {
        int k0 = kc * 32;
        float vals[16];
        if (k0 + cb < 128) load16(pAgg + k0, vals);
        else               load16(pH + k0 - 128, vals);
        packStore(&As[r][cb], vals);
        stageB(wOut, OUT_K, 128, k0, &Bs[0][0], tid);
        __syncthreads();
        Frag16 a = loadA(&As[0][0], wv * 16, lane);
        Frag16 bf = loadB(&Bs[0][0], 0, lane);
#pragma unroll
        for (int t = 0; t < 8; ++t) {
            Frag16 bn;
            if (t + 1 < 8) bn = loadB(&Bs[0][0], (t + 1) * 16, lane);
            acc[t] = __builtin_amdgcn_wmma_f32_16x16x32_bf16(
                false, a.v, false, bf.v, (short)0, acc[t], false, false);
            if (t + 1 < 8) bf = bn;
        }
        __syncthreads();
    }
    // epilogue: bias + relu + layernorm (half-wave shuffles reduce each row independently)
#pragma unroll
    for (int r8 = 0; r8 < 8; ++r8) {
        float vals[8];
        float s = 0.f;
#pragma unroll
        for (int t = 0; t < 8; ++t) {
            int col = t * 16 + (lane & 15);
            float x = acc[t][r8] + ob[col];
            x = (x > 0.f) ? x : 0.f;
            vals[t] = x;
            s += x;
        }
        s += __shfl_xor(s, 1, 32); s += __shfl_xor(s, 2, 32);
        s += __shfl_xor(s, 4, 32); s += __shfl_xor(s, 8, 32);
        float mean = s * (1.f / 128.f);
        float vs = 0.f;
#pragma unroll
        for (int t = 0; t < 8; ++t) { float dlt = vals[t] - mean; vs += dlt * dlt; }
        vs += __shfl_xor(vs, 1, 32); vs += __shfl_xor(vs, 2, 32);
        vs += __shfl_xor(vs, 4, 32); vs += __shfl_xor(vs, 8, 32);
        float rstd = rsqrtf(vs * (1.f / 128.f) + 1e-5f);
        int row = mBase + wv * 16 + r8 + 8 * (lane >> 4);
#pragma unroll
        for (int t = 0; t < 8; ++t) {
            int col = t * 16 + (lane & 15);
            out[(size_t)row * 128 + col] = (vals[t] - mean) * rstd * g[col] + bb[col];
        }
    }
}

// ---------- launch ----------
extern "C" void kernel_launch(void* const* d_in, const int* in_sizes, int n_in,
                              void* d_out, int out_size, void* d_ws, size_t ws_size,
                              hipStream_t stream) {
    (void)in_sizes; (void)n_in; (void)out_size; (void)ws_size;
    const float* h_src     = (const float*)d_in[0];
    const float* edge_feat = (const float*)d_in[1];
    const float* dtv       = (const float*)d_in[2];
    const float* w_q_w     = (const float*)d_in[3];
    const float* w_q_b     = (const float*)d_in[4];
    const float* w_k_w     = (const float*)d_in[5];
    const float* w_k_b     = (const float*)d_in[6];
    const float* w_v_w     = (const float*)d_in[7];
    const float* w_v_b     = (const float*)d_in[8];
    const float* w_out_w   = (const float*)d_in[9];
    const float* w_out_b   = (const float*)d_in[10];
    const float* ln_g      = (const float*)d_in[11];
    const float* ln_b      = (const float*)d_in[12];
    const float* time_w    = (const float*)d_in[13];
    const float* time_b    = (const float*)d_in[14];
    const int*   edge_dst  = (const int*)d_in[15];

    char* ws = (char*)d_ws;
    u16*   wKV  = (u16*)(ws + OFF_WKV);
    u16*   wQ   = (u16*)(ws + OFF_WQ);
    u16*   wOut = (u16*)(ws + OFF_WOUT);
    float* qc   = (float*)(ws + OFF_QCONST);
    int*   rowp = (int*)(ws + OFF_ROWP);
    float* Qb   = (float*)(ws + OFF_Q);
    float* aggb = (float*)(ws + OFF_AGG);
    u16*   KVb  = (u16*)(ws + OFF_KV);

    prep_weights<<<576, 256, 0, stream>>>(w_q_w, w_k_w, w_v_w, w_out_w, wQ, wKV, wOut);
    prep_qconst<<<1, 128, 0, stream>>>(w_q_w, time_b, qc);
    build_rowptr<<<N_EDGE / 256, 256, 0, stream>>>(edge_dst, rowp);
    gemm_q<<<N_DST / 128, 256, 0, stream>>>(h_src, wQ, w_q_b, qc, Qb);
    gemm_kv<<<N_EDGE / 128, 256, 0, stream>>>(h_src, edge_feat, dtv, time_w, time_b,
                                              wKV, w_k_b, w_v_b, KVb);
    attn<<<N_DST / 8, 256, 0, stream>>>(Qb, KVb, rowp, aggb);
    gemm_out<<<N_DST / 128, 256, 0, stream>>>(aggb, h_src, wOut, w_out_b, ln_g, ln_b,
                                              (float*)d_out);
}
